// GenNCA_v3_33191507264154
// MI455X (gfx1250) — matrix operand order repaired and checked by reading, not compiled, not run on hardware
//
#include <hip/hip_runtime.h>

typedef __attribute__((ext_vector_type(16))) _Float16 v16h;
typedef __attribute__((ext_vector_type(8)))  float    v8f;
typedef __attribute__((ext_vector_type(4)))  unsigned int v4u;
typedef __attribute__((ext_vector_type(8)))  unsigned int v8u;

struct __align__(16) h8 { _Float16 h[8]; };

#define SDIM  256
#define CH    16
#define HID   128
#define BATCH 8
#define NSTEP 10
#define FIRE  0.5f
#define EPSV  1e-5f

// ---------------- workspace byte offsets ----------------
static const size_t OFF_X0    = 0;                       // 33,554,432  f32 state ping
static const size_t OFF_X1    = 33554432;                // 33,554,432  f32 state pong
static const size_t OFF_DXH   = 67108864;                // 134,217,728 f16 [b][65536 pix][128 ch]
static const size_t OFF_HCAT  = 201326592;               // 16,640      f32 [8][520]
static const size_t OFF_CONVW = 201343232;               // 25,088      f32 [b*16+c][49]
static const size_t OFF_W1H   = 201368320;               // 65,536      f16 [b][128][32]
static const size_t OFF_W2H   = 201433856;               // 32,768      f16 [b][16][128]
static const size_t OFF_STAT  = 201466624;               // colsum 8*32 f32 + M 8*1024 f32
static const size_t OFF_BNP   = 201500672;               // 1,024       f32 rsg[128],shift[128]

// ---------------- KA dynamic-LDS layout ----------------
#define KA_PATCH_OFF 0         // f32 patch [8][262][16] = 134144 B ; reused as Mpart [8][1024] f32
#define KA_X2_OFF    134144    // f16 x2t [512][40]      =  40960 B (pitch 40: 16B-aligned, 20-bank stride)
#define KA_W1_OFF    175104    // f16 w1l [128][40]      =  10240 B (filled by TDM with LDS padding)
#define KA_WT_OFF    185344    // f32 wtl [49][16]       =   3136 B
#define KA_CS_OFF    188480    // f32 csS [256][32]      =  32768 B
#define KA_LDS_BYTES 221248

// =======================================================
// init: x [B,S,S,C] channels-last  ->  X [b][c][i][j]  (X[b,c,i,j] = x[b,j,i,c])
__global__ void __launch_bounds__(256) k_init(const float* __restrict__ x,
                                              float* __restrict__ X0) {
  int i = blockIdx.x * 256 + threadIdx.x;        // 8,388,608 elements
  int jj = i & 255, ii = (i >> 8) & 255, c = (i >> 16) & 15, b = i >> 20;
  X0[i] = x[((((size_t)b * SDIM + jj) * SDIM + ii) << 4) + c];
}

__global__ void k_zero(float* __restrict__ g) {
  int i = blockIdx.x * 256 + threadIdx.x;
  if (i < 8448) g[i] = 0.f;                      // colsum (256) + M (8192)
}

// =======================================================
// Hypernetwork stage 1: emb, h1, h2 -> hcat [8][520]
__global__ void __launch_bounds__(256) kh_hyper1(
    const float* __restrict__ xv, const float* __restrict__ bb,
    const float* __restrict__ W01, const float* __restrict__ b01,
    const float* __restrict__ W02, const float* __restrict__ b02,
    float* __restrict__ hcat) {
  __shared__ float embL[64];
  __shared__ float h1L[512];
  const int tid = threadIdx.x;
  if (tid < 64) embL[tid] = xv[tid] + bb[tid];
  __syncthreads();
  for (int idx = tid; idx < 512; idx += 256) {
    int s = idx >> 6, o = idx & 63;
    float a = b01[o];
#pragma unroll
    for (int k = 0; k < 8; ++k) a = fmaf(embL[s * 8 + k], W01[o * 8 + k], a);
    h1L[idx] = fmaxf(a, 0.f);
  }
  __syncthreads();
  for (int idx = tid; idx < 4096; idx += 256) {
    int s = idx >> 9, o = idx & 511;
    float a = b02[o];
    for (int k = 0; k < 64; ++k) a = fmaf(h1L[s * 64 + k], W02[o * 72 + k], a);
#pragma unroll
    for (int k = 0; k < 8; ++k) a = fmaf(embL[s * 8 + k], W02[o * 72 + 64 + k], a);
    hcat[s * 520 + o] = fmaxf(a, 0.f);
  }
  if (tid < 64) hcat[(tid >> 3) * 520 + 512 + (tid & 7)] = embL[tid];
}

// Hypernetwork stage 2: gw routed into conv_w(f32) / W1(f16) / W2(f16)
__global__ void __launch_bounds__(256) kh_hyper2(
    const float* __restrict__ hcat, const float* __restrict__ W05,
    const float* __restrict__ b05, float* __restrict__ convw,
    _Float16* __restrict__ w1h, _Float16* __restrict__ w2h) {
  __shared__ float hL[8 * 520];
  const int tid = threadIdx.x;
  for (int i = tid; i < 8 * 520; i += 256) hL[i] = hcat[i];
  __syncthreads();
  const int out = blockIdx.x * 256 + tid;
  if (out >= 6928) return;
  float acc[8];
#pragma unroll
  for (int s = 0; s < 8; ++s) acc[s] = b05[out];
  const float* wr = W05 + (size_t)out * 520;
  for (int k = 0; k < 520; ++k) {
    float w = wr[k];
#pragma unroll
    for (int s = 0; s < 8; ++s) acc[s] = fmaf(w, hL[s * 520 + k], acc[s]);
  }
  if (out < 784) {
    int c = out / 49, r = out % 49;
#pragma unroll
    for (int s = 0; s < 8; ++s) convw[(s * CH + c) * 49 + r] = acc[s];
  } else if (out < 784 + 4096) {
    int q = out - 784;
#pragma unroll
    for (int s = 0; s < 8; ++s) w1h[s * HID * 32 + q] = (_Float16)acc[s];
  } else {
    int q = out - 4880;
#pragma unroll
    for (int s = 0; s < 8; ++s) w2h[s * CH * HID + q] = (_Float16)acc[s];
  }
}

// =======================================================
// KA: TDM-staged W1 -> halo patch -> depthwise 7x7 -> WMMA GEMM1 (32->128)
//     + colsum[32] and second-moment M = X2*X2^T (via WMMA) for BN stats
__global__ void __launch_bounds__(256) ka_perceive_gemm1(
    const float* __restrict__ Xin, const float* __restrict__ convw,
    const _Float16* __restrict__ w1h, _Float16* __restrict__ dxh,
    float* __restrict__ colsumG, float* __restrict__ MG) {
  extern __shared__ char smem[];
  float*    patch = (float*)(smem + KA_PATCH_OFF);
  float*    Mpart = (float*)(smem + KA_PATCH_OFF);   // alias: valid after conv phase
  _Float16* x2t   = (_Float16*)(smem + KA_X2_OFF);
  _Float16* w1l   = (_Float16*)(smem + KA_W1_OFF);
  float*    wtl   = (float*)(smem + KA_WT_OFF);
  float*    csS   = (float*)(smem + KA_CS_OFF);

  const int tid  = threadIdx.x;
  const int b    = blockIdx.y;
  const int row0 = blockIdx.x * 2;

  // ---- TDM: DMA W1 (4096 f16, contiguous) into LDS with hardware row padding:
  //      pad_interval=3 (every 16 DWORDs = 32 halves), pad_amount=3 (4 DWORDs = 8 halves)
  //      -> reproduces the pitch-40-halves layout.  (D# per ISA 08_async_tensor §8.3/8.4)
  if (tid < 32) {                                    // wave 0 issues; TDM ignores EXEC
    const size_t ga = (size_t)(const void*)(w1h + (size_t)b * HID * 32);
    v4u g0;
    g0[0] = 1u;                                      // count=1 (valid user descriptor)
    g0[1] = (unsigned)KA_W1_OFF;                     // lds_addr (bytes)
    g0[2] = (unsigned)(ga & 0xffffffffu);            // global_addr[31:0]
    g0[3] = ((unsigned)(ga >> 32) & 0x01FFFFFFu) | (2u << 30);  // addr[56:32] | type=2
    v8u g1;
    g1[0] = (1u << 16) | (1u << 20) | (3u << 22) | (3u << 25);  // data=2B, pad on
    g1[1] = (4096u & 0xffffu) << 16;                 // tensor_dim0 = 4096 (low 16)
    g1[2] = (1u << 16);                              // dim0 hi=0 ; tensor_dim1 = 1
    g1[3] = (4096u << 16);                           // dim1 hi=0 ; tile_dim0 = 4096
    g1[4] = 0u;                                      // tile_dim1/2 unused
    g1[5] = 4096u;                                   // tensor_dim0_stride
    g1[6] = 0u;
    g1[7] = 0u;
    asm volatile("tensor_load_to_lds %0, %1" :: "s"(g0), "s"(g1) : "memory");
    __builtin_amdgcn_s_wait_tensorcnt(0);
  }

  for (int i = tid; i < 49 * CH; i += 256) {         // wtl[r][ch]
    int r = i >> 4, ch = i & 15;
    wtl[i] = convw[(b * CH + ch) * 49 + r];
  }
  for (int ch = 0; ch < CH; ++ch) {                  // 8 x 262 halo patch, zero-padded
    const float* src = Xin + (((size_t)(b * CH + ch)) << 16);
    for (int l = tid; l < 8 * 262; l += 256) {
      int r = l / 262, c = l % 262;
      int gi = row0 - 3 + r, gj = c - 3;
      float v = 0.f;
      if (gi >= 0 && gi < SDIM && gj >= 0 && gj < SDIM) v = src[gi * SDIM + gj];
      patch[l * CH + ch] = v;
    }
  }
  __syncthreads();

  // depthwise 7x7, 2 pixels/thread; emit x2t[pix][32] f16 + per-thread colsum partials
  float sxy[32];
#pragma unroll
  for (int k = 0; k < 32; ++k) sxy[k] = 0.f;
  for (int pp = 0; pp < 2; ++pp) {
    const int p = tid + pp * 256;
    const int lr = p >> 8, lc = p & 255;
    float acc[CH];
#pragma unroll
    for (int c = 0; c < CH; ++c) acc[c] = 0.f;
    for (int r = 0; r < 49; ++r) {
      const int ki = r / 7, kj = r % 7;
      const float* pr = &patch[((lr + ki) * 262 + (lc + kj)) * CH];
      const float* wr = &wtl[r * CH];
#pragma unroll
      for (int c = 0; c < CH; ++c) acc[c] = fmaf(pr[c], wr[c], acc[c]);
    }
    const float* ctr = &patch[((lr + 3) * 262 + (lc + 3)) * CH];
    h8 pk[4];
#pragma unroll
    for (int c = 0; c < CH; ++c) {
      float xv = ctr[c], yv = acc[c];
      sxy[c] += xv;
      sxy[16 + c] += yv;
      ((_Float16*)pk)[c] = (_Float16)xv;
      ((_Float16*)pk)[16 + c] = (_Float16)yv;
    }
    h8* dst = (h8*)&x2t[p * 40];                     // 4 x ds_store_b128
    dst[0] = pk[0]; dst[1] = pk[1]; dst[2] = pk[2]; dst[3] = pk[3];
  }
#pragma unroll
  for (int k = 0; k < 32; ++k) csS[tid * 32 + k] = sxy[k];
  __syncthreads();

  const int wv = tid >> 5, lane = tid & 31;
  const int hi = lane >> 4, lo = lane & 15;

  // ---- GEMM1: D[128 x 512] = W1[128 x 32] * X2[32 x 512] ----
  // phase 1: 8 independent accumulators (kills WMMA->VALU hazard NOPs);
  // phase 2: convert + packed b128 stores.
  for (int t4 = 0; t4 < 4; ++t4) {
    const int n = wv * 64 + t4 * 16 + lo;
    const int gpix = row0 * SDIM + n;
    v16h bfrag;                                      // B 32x16: K = hi*8(+16)+h
    {
      const h8* bp = (const h8*)&x2t[n * 40 + hi * 8];
      h8 b0 = bp[0], b1 = bp[2];
#pragma unroll
      for (int h = 0; h < 8; ++h) { bfrag[h] = b0.h[h]; bfrag[h + 8] = b1.h[h]; }
    }
    v8f cc[8];
#pragma unroll
    for (int mt = 0; mt < 8; ++mt) {
      v16h afrag;                                    // A 16x32: row lo, K = hi*8(+16)+h
      {
        const h8* a0p = (const h8*)&w1l[(mt * 16 + lo) * 40 + hi * 8];
        const h8* a1p = (const h8*)&w1l[(mt * 16 + lo) * 40 + 16 + hi * 8];
        h8 a0 = *a0p, a1 = *a1p;
#pragma unroll
        for (int h = 0; h < 8; ++h) { afrag[h] = a0.h[h]; afrag[h + 8] = a1.h[h]; }
      }
      v8f z = {};
      cc[mt] = __builtin_amdgcn_wmma_f32_16x16x32_f16(false, afrag, false, bfrag,
                                                      (short)0, z, false, false);
    }
    _Float16* dbase = dxh + ((size_t)((b << 16) + gpix)) * 128;
#pragma unroll
    for (int mt = 0; mt < 8; ++mt) {
      h8 pk;                                         // D row mt*16+hi*8+v, col n
#pragma unroll
      for (int v = 0; v < 8; ++v) pk.h[v] = (_Float16)cc[mt][v];
      *(h8*)(dbase + mt * 16 + hi * 8) = pk;         // one global_store_b128
    }
  }

  // ---- second moment M = X2 * X2^T (32x32, K=512): wave wv covers chunks wv, wv+8 ----
  v8f macc[4] = {v8f{}, v8f{}, v8f{}, v8f{}};
#pragma unroll
  for (int cidx = 0; cidx < 2; ++cidx) {
    const int pb = (wv + cidx * 8) * 32;
    v16h af[2], bf[2];
#pragma unroll
    for (int mr = 0; mr < 2; ++mr) {
      const int rr = mr * 16 + lo;
#pragma unroll
      for (int h = 0; h < 8; ++h) {                  // A: rows=channels, K=pixels
        af[mr][h]     = x2t[(pb + hi * 8 + h) * 40 + rr];
        af[mr][h + 8] = x2t[(pb + 16 + hi * 8 + h) * 40 + rr];
      }
#pragma unroll
      for (int h = 0; h < 16; ++h)                   // B: cols=channels, K=pixels
        bf[mr][h] = af[mr][h];
    }
    macc[0] = __builtin_amdgcn_wmma_f32_16x16x32_f16(false, af[0], false, bf[0], (short)0, macc[0], false, false);
    macc[1] = __builtin_amdgcn_wmma_f32_16x16x32_f16(false, af[0], false, bf[1], (short)0, macc[1], false, false);
    macc[2] = __builtin_amdgcn_wmma_f32_16x16x32_f16(false, af[1], false, bf[0], (short)0, macc[2], false, false);
    macc[3] = __builtin_amdgcn_wmma_f32_16x16x32_f16(false, af[1], false, bf[1], (short)0, macc[3], false, false);
  }
  {
    float* mp = Mpart + wv * 1024;                   // wave-private partial (patch reused)
#pragma unroll
    for (int tile = 0; tile < 4; ++tile) {
      const int mr = tile >> 1, mc = tile & 1;
#pragma unroll
      for (int v = 0; v < 8; ++v)
        mp[(mr * 16 + hi * 8 + v) * 32 + mc * 16 + lo] = macc[tile][v];
    }
  }
  __syncthreads();

  // ---- block-level reductions -> global atomics ----
  if (tid < 32) {
    float s = 0.f;
    for (int tt = 0; tt < 256; ++tt) s += csS[tt * 32 + tid];
    unsafeAtomicAdd(&colsumG[b * 32 + tid], s);
  }
#pragma unroll
  for (int e = 0; e < 4; ++e) {
    const int idx = tid * 4 + e;
    float s = 0.f;
#pragma unroll
    for (int w = 0; w < 8; ++w) s += Mpart[w * 1024 + idx];
    unsafeAtomicAdd(&MG[b * 1024 + idx], s);
  }
}

// =======================================================
// KB: BN stats from colsum/M:  sum = W1*colsum ; sumsq = w1^T M w1
__global__ void __launch_bounds__(128) kb_bnstats(
    const _Float16* __restrict__ w1h, const float* __restrict__ colsumG,
    const float* __restrict__ MG, const float* __restrict__ gamma,
    const float* __restrict__ beta, float* __restrict__ bnp) {
  __shared__ float Ms[8 * 1024];
  __shared__ float cs[8 * 32];
  const int tid = threadIdx.x;
  for (int i = tid; i < 8192; i += 128) Ms[i] = MG[i];
  for (int i = tid; i < 256; i += 128) cs[i] = colsumG[i];
  __syncthreads();
  const int chn = tid;
  float s = 0.f, s2 = 0.f;
  for (int b = 0; b < 8; ++b) {
    float wv[32];
#pragma unroll
    for (int k = 0; k < 32; ++k) wv[k] = (float)w1h[(b * HID + chn) * 32 + k];
#pragma unroll
    for (int k = 0; k < 32; ++k) s += wv[k] * cs[b * 32 + k];
    for (int j = 0; j < 32; ++j) {
      float mj = 0.f;
#pragma unroll
      for (int k = 0; k < 32; ++k) mj = fmaf(Ms[b * 1024 + j * 32 + k], wv[k], mj);
      s2 = fmaf(wv[j], mj, s2);
    }
  }
  const float inv_n = 1.0f / (float)(BATCH * SDIM * SDIM);
  float mean = s * inv_n;
  float var  = s2 * inv_n - mean * mean;
  float rsg  = rsqrtf(var + EPSV) * gamma[chn];
  bnp[chn]       = rsg;
  bnp[HID + chn] = beta[chn] - mean * rsg;
}

// =======================================================
// KC: normalize+ReLU -> WMMA GEMM2 (16x128, K=4x32) -> mask+residual
//     last step writes d_out directly in channels-last layout
__global__ void __launch_bounds__(256) kc_bn_gemm2_update(
    const _Float16* __restrict__ dxh, const _Float16* __restrict__ w2h,
    const float* __restrict__ bnp, const float* __restrict__ Xin,
    const float* __restrict__ rmask, float* __restrict__ Xout,
    float* __restrict__ outFinal, int t) {
  __shared__ _Float16 w2l[CH * 136];                 // pitch 136: 16B-aligned, 4-bank stride
  __shared__ float bnl[2 * HID];
  const int tid  = threadIdx.x;
  const int b    = blockIdx.y;
  const int row0 = blockIdx.x * 2;
  for (int i = tid; i < CH * HID; i += 256) {
    int r = i >> 7, k = i & 127;
    w2l[r * 136 + k] = w2h[b * CH * HID + i];
  }
  if (tid < 2 * HID) bnl[tid] = bnp[tid];
  __syncthreads();

  const int wv = tid >> 5, lane = tid & 31;
  const int hi = lane >> 4, lo = lane & 15;
  for (int t4 = 0; t4 < 4; ++t4) {
    const int n = wv * 64 + t4 * 16 + lo;
    const int gpix = row0 * SDIM + n;
    const _Float16* dbase = dxh + ((size_t)((b << 16) + gpix)) * 128;
    v8f c = {};
#pragma unroll
    for (int kc = 0; kc < 4; ++kc) {
      v16h bfrag;                                    // K rows = channels kc*32 + hi*8(+16) + h
      {
        const h8* bp = (const h8*)(dbase + kc * 32 + hi * 8);
        h8 b0 = bp[0], b1 = bp[2];                   // two global_load_b128
#pragma unroll
        for (int h = 0; h < 8; ++h) {
          const int c0 = kc * 32 + hi * 8 + h;
          const int c1 = c0 + 16;
          float f0 = fmaf((float)b0.h[h], bnl[c0], bnl[HID + c0]);
          float f1 = fmaf((float)b1.h[h], bnl[c1], bnl[HID + c1]);
          bfrag[h]     = (_Float16)fmaxf(f0, 0.f);
          bfrag[h + 8] = (_Float16)fmaxf(f1, 0.f);
        }
      }
      v16h afrag;                                    // A = W2[16 x 128] chunk kc
      {
        const h8* a0p = (const h8*)&w2l[lo * 136 + kc * 32 + hi * 8];
        const h8* a1p = (const h8*)&w2l[lo * 136 + kc * 32 + 16 + hi * 8];
        h8 a0 = *a0p, a1 = *a1p;
#pragma unroll
        for (int h = 0; h < 8; ++h) { afrag[h] = a0.h[h]; afrag[h + 8] = a1.h[h]; }
      }
      c = __builtin_amdgcn_wmma_f32_16x16x32_f16(false, afrag, false, bfrag,
                                                 (short)0, c, false, false);
    }
    // stochastic firing: mask[t][b][j][i], i = gpix>>8, j = gpix&255
    const float mv =
        rmask[(((size_t)((t * BATCH + b) * SDIM + (gpix & 255))) << 8) + (gpix >> 8)];
    const float keep = (mv > FIRE) ? 1.f : 0.f;
    if (outFinal) {                                  // channels-last: out[b][j][i][c]
      float* op = outFinal +
          ((size_t)((b << 16) + ((gpix & 255) << 8) + (gpix >> 8))) * CH + hi * 8;
#pragma unroll
      for (int v = 0; v < 8; ++v) {
        const int oc = hi * 8 + v;
        op[v] = Xin[(((size_t)(b * CH + oc)) << 16) + gpix] + keep * c[v];
      }
    } else {
#pragma unroll
      for (int v = 0; v < 8; ++v) {
        const int oc = hi * 8 + v;
        const size_t idx = (((size_t)(b * CH + oc)) << 16) + gpix;
        Xout[idx] = Xin[idx] + keep * c[v];
      }
    }
  }
}

// =======================================================
extern "C" void kernel_launch(void* const* d_in, const int* in_sizes, int n_in,
                              void* d_out, int out_size, void* d_ws, size_t ws_size,
                              hipStream_t stream) {
  const float* x     = (const float*)d_in[0];
  const float* x_vec = (const float*)d_in[1];
  const float* biasb = (const float*)d_in[2];
  const float* W01   = (const float*)d_in[3];
  const float* b01   = (const float*)d_in[4];
  const float* W02   = (const float*)d_in[5];
  const float* b02   = (const float*)d_in[6];
  const float* W05   = (const float*)d_in[7];
  const float* b05   = (const float*)d_in[8];
  const float* gamma = (const float*)d_in[9];
  const float* beta  = (const float*)d_in[10];
  const float* rmask = (const float*)d_in[11];
  // d_in[12] = steps (fixed 10 in reference setup)

  char* ws = (char*)d_ws;
  float*    X0      = (float*)(ws + OFF_X0);
  float*    X1      = (float*)(ws + OFF_X1);
  _Float16* dxh     = (_Float16*)(ws + OFF_DXH);
  float*    hcat    = (float*)(ws + OFF_HCAT);
  float*    convw   = (float*)(ws + OFF_CONVW);
  _Float16* w1h     = (_Float16*)(ws + OFF_W1H);
  _Float16* w2h     = (_Float16*)(ws + OFF_W2H);
  float*    colsumG = (float*)(ws + OFF_STAT);
  float*    MG      = (float*)(ws + OFF_STAT + 1024);
  float*    bnp     = (float*)(ws + OFF_BNP);

  hipFuncSetAttribute((const void*)ka_perceive_gemm1,
                      hipFuncAttributeMaxDynamicSharedMemorySize, KA_LDS_BYTES);

  dim3 blk(256);
  k_init<<<32768, blk, 0, stream>>>(x, X0);
  kh_hyper1<<<1, blk, 0, stream>>>(x_vec, biasb, W01, b01, W02, b02, hcat);
  kh_hyper2<<<28, blk, 0, stream>>>(hcat, W05, b05, convw, w1h, w2h);

  for (int t = 0; t < NSTEP; ++t) {
    const float* Xi = (t & 1) ? X1 : X0;
    float*       Xo = (t & 1) ? X0 : X1;
    k_zero<<<33, blk, 0, stream>>>(colsumG);         // zeros colsum + M
    ka_perceive_gemm1<<<dim3(SDIM / 2, BATCH), blk, KA_LDS_BYTES, stream>>>(
        Xi, convw, w1h, dxh, colsumG, MG);
    kb_bnstats<<<1, dim3(128), 0, stream>>>(w1h, colsumG, MG, gamma, beta, bnp);
    kc_bn_gemm2_update<<<dim3(SDIM / 2, BATCH), blk, 0, stream>>>(
        dxh, w2h, bnp, Xi, rmask, Xo,
        (t == NSTEP - 1) ? (float*)d_out : nullptr, t);
  }
}
